// Cross_Attn_21973052686357
// MI455X (gfx1250) — compile-verified, matrix operand order checked by reading, not run on hardware
//
#include <hip/hip_runtime.h>
#include <hip/hip_bf16.h>

// ---- problem constants (match reference) ----
#define B_    4
#define CQ_   384
#define CK_   384
#define COUT_ 384
#define N_    2048
#define K_    16
#define G_    32
#define CPG_  (COUT_/G_)          // 12 channels per group
#define EPS_  1e-5f
#define SCALE_ 0.05103103630798288f   // 384^-0.5
#define LB_   (CK_ + 8)               // bf16 LDS row stride [k][c]

typedef __bf16 v16bf __attribute__((ext_vector_type(16)));
typedef __bf16 v8bf  __attribute__((ext_vector_type(8)));
typedef float  v8f   __attribute__((ext_vector_type(8)));
typedef unsigned int v4u __attribute__((ext_vector_type(4)));
typedef int    v8i   __attribute__((ext_vector_type(8)));
typedef int    v4i   __attribute__((ext_vector_type(4)));

// Build a 16-element bf16 fragment from two contiguous 16-byte chunks:
// elements 0..7 at p[0..7], elements 8..15 at p[16..23]  (matches the
// 16-bit A/B wave32 striping: lane half 0 -> K {0..7,16..23}, half 1 -> {8..15,24..31})
__device__ __forceinline__ v16bf pack16(const __bf16* p) {
    const v8bf* q = (const v8bf*)p;
    v8bf lo = q[0];
    v8bf hi = q[2];              // +16 elements
    v16bf r;
#pragma unroll
    for (int j = 0; j < 8; ++j) { r[j] = lo[j]; r[j + 8] = hi[j]; }
    return r;
}

// ---------------------------------------------------------------------------
// TDM: DMA a 384(c) x 16(k) fp32 tile (row stride N*K floats) into LDS,
// with TDM padding of 1 DWORD after every 16 DWORDs -> LDS row stride 17.
// D# layout per CDNA5 ISA ch.8 (group0 128b, group1 256b; groups 2/3 unused
// for 2D tiles).
// ---------------------------------------------------------------------------
__device__ __forceinline__ void tdm_load_tile(const float* gsrc, unsigned lds_off) {
    unsigned long long ga = (unsigned long long)gsrc;
    v4u g0;
    g0.x = 1u;                                               // count=1 (valid), user mode
    g0.y = lds_off;                                          // lds_addr (bytes)
    g0.z = (unsigned)(ga & 0xffffffffull);                   // global_addr[31:0]
    g0.w = (unsigned)((ga >> 32) & 0x1ffffffull) | (2u << 30); // addr[56:32] | type=2
    v8i g1;
    g1[0] = (2 << 16) | (1 << 20) | (3 << 22);  // data_size=4B | pad_enable | pad_interval=16DW (amount=1DW)
    g1[1] = (K_ << 16);                         // atomic_barrier_addr=0 | tensor_dim0.lo16 = 16
    g1[2] = (CK_ << 16);                        // tensor_dim0.hi16=0 | tensor_dim1.lo16 = 384
    g1[3] = (K_ << 16);                         // tensor_dim1.hi16=0 | tile_dim0 = 16
    g1[4] = CK_;                                // tile_dim1 = 384 | tile_dim2 = 0
    g1[5] = N_ * K_;                            // tensor_dim0_stride = 32768
    g1[6] = 0;                                  // stride.hi16 | tensor_dim1_stride.lo16
    g1[7] = 0;                                  // tensor_dim1_stride.hi32
    v4i z4 = {0, 0, 0, 0};
#if __clang_major__ >= 23
    v8i z8 = {0, 0, 0, 0, 0, 0, 0, 0};
    __builtin_amdgcn_tensor_load_to_lds(g0, g1, z4, z4, z8, 0);
#else
    __builtin_amdgcn_tensor_load_to_lds(g0, g1, z4, z4, 0);
#endif
}

// ---------------------------------------------------------------------------
// Kernel 1a: Mt[c2,c1] = sum_o Wq[o,c1]*Wk[o,c2]  (stored transposed, bf16)
// ---------------------------------------------------------------------------
__global__ void k_prep_M(const float* __restrict__ Wq, const float* __restrict__ Wk,
                         __bf16* __restrict__ Mt) {
    int i = blockIdx.x * blockDim.x + threadIdx.x;   // i = c2*CQ_ + c1
    if (i >= CQ_ * CK_) return;
    int c2 = i / CQ_, c1 = i % CQ_;
    float acc = 0.f;
    for (int o = 0; o < COUT_; ++o)
        acc += Wq[o * CQ_ + c1] * Wk[o * CK_ + c2];
    Mt[i] = (__bf16)acc;
}

// ---------------------------------------------------------------------------
// Kernel 1b: Wv->bf16, bqk = Wk^T bq, wqb = Wq^T bk, cbb = bq.bk, zero stats
// ---------------------------------------------------------------------------
__global__ void k_prep_misc(const float* __restrict__ Wq, const float* __restrict__ Wk,
                            const float* __restrict__ Wv, const float* __restrict__ bq,
                            const float* __restrict__ bk,
                            __bf16* __restrict__ Wvbf, float* __restrict__ bqk,
                            float* __restrict__ wqb, float* __restrict__ cbb,
                            float* __restrict__ stats) {
    int i = blockIdx.x * blockDim.x + threadIdx.x;
    if (i < COUT_ * CK_) Wvbf[i] = (__bf16)Wv[i];
    if (i < CK_) {
        float a = 0.f;
        for (int o = 0; o < COUT_; ++o) a += bq[o] * Wk[o * CK_ + i];
        bqk[i] = a;
    }
    if (i < CQ_) {
        float a = 0.f;
        for (int o = 0; o < COUT_; ++o) a += Wq[o * CQ_ + i] * bk[o];
        wqb[i] = a;
    }
    if (i == 0) {
        float a = 0.f;
        for (int o = 0; o < COUT_; ++o) a += bq[o] * bk[o];
        *cbb = a;
    }
    if (i < B_ * G_ * 2) stats[i] = 0.f;
}

// ---------------------------------------------------------------------------
// Kernel 1c: featT[b][n][c] = feat[b][c][n] (bf16) so A-fragments are b128 loads
// ---------------------------------------------------------------------------
__global__ void k_prep_featT(const float* __restrict__ feat, __bf16* __restrict__ featT) {
    int i = blockIdx.x * blockDim.x + threadIdx.x;
    if (i >= B_ * N_ * CQ_) return;
    int c = i % CQ_;
    int n = (i / CQ_) % N_;
    int b = i / (CQ_ * N_);
    featT[i] = (__bf16)feat[((size_t)b * CQ_ + c) * N_ + n];
}

// ---------------------------------------------------------------------------
// Kernel 2: qk[b,n,c] = sum_c' featT[b,n,c'] * Mt[c,c'] + bqk[c]   (bf16 out)
// One wave per 16n x 16c tile; 12 WMMA k-steps; all operands via b128 loads.
// ---------------------------------------------------------------------------
__global__ __launch_bounds__(256) void k_qk(const __bf16* __restrict__ featT,
                                            const __bf16* __restrict__ Mt,
                                            const float* __restrict__ bqk,
                                            __bf16* __restrict__ qkb) {
    int lane = threadIdx.x & 31;
    int wave = threadIdx.x >> 5;
    int wid  = blockIdx.x * (blockDim.x >> 5) + wave;
    int tilesPerB = (N_ / 16) * (CK_ / 16);     // 128*24 = 3072
    int b   = wid / tilesPerB;
    int rem = wid % tilesPerB;
    int n0  = (rem / (CK_ / 16)) * 16;
    int c0  = (rem % (CK_ / 16)) * 16;
    int half = lane >> 4, lrow = lane & 15;

    const __bf16* arow = featT + ((size_t)b * N_ + n0 + lrow) * CQ_ + half * 8;
    const __bf16* brow = Mt + (size_t)(c0 + lrow) * CQ_ + half * 8;
    v8f acc = {};
#pragma unroll 2
    for (int cs = 0; cs < CQ_ / 32; ++cs) {
        v16bf a  = pack16(arow + cs * 32);
        v16bf bm = pack16(brow + cs * 32);
        acc = __builtin_amdgcn_wmma_f32_16x16x32_bf16(false, a, false, bm,
                                                      (short)0, acc, false, false);
    }
#pragma unroll
    for (int r = 0; r < 8; ++r) {
        int n = n0 + r + half * 8;
        int c = c0 + lrow;
        qkb[((size_t)b * N_ + n) * CK_ + c] = (__bf16)(acc[r] + bqk[c]);
    }
}

// ---------------------------------------------------------------------------
// Kernel 3: s0[b,n] = sum_c feat[b,c,n]*wqb[c] + cbb   (tiny)
// ---------------------------------------------------------------------------
__global__ void k_s0(const float* __restrict__ feat, const float* __restrict__ wqb,
                     const float* __restrict__ cbb, float* __restrict__ s0) {
    int i = blockIdx.x * blockDim.x + threadIdx.x;
    if (i >= B_ * N_) return;
    int b = i / N_, n = i % N_;
    const float* fb = feat + (size_t)b * CQ_ * N_ + n;
    float a = *cbb;
    for (int c = 0; c < CQ_; ++c) a += fb[(size_t)c * N_] * wqb[c];
    s0[i] = a;
}

__device__ __forceinline__ float red16(float v) {
#pragma unroll
    for (int m = 8; m >= 1; m >>= 1) v += __shfl_xor(v, m, 16);
    return v;
}

// ---------------------------------------------------------------------------
// Kernel 4: per (b,n): TDM-stage gf[:,n,:] (384x16 fp32) into LDS; convert
// once into bf16 [k][c]; compute masked softmax p[b,n,k] and v = Wv*gf + bv
// via WMMA (8 waves x 3 o-tiles), accumulating per-(b,g) GN sum/sumsq.
// ---------------------------------------------------------------------------
__global__ __launch_bounds__(256) void k_stats(const float* __restrict__ gf,
                                               const __bf16* __restrict__ Wvbf,
                                               const float* __restrict__ bv,
                                               const __bf16* __restrict__ qkb,
                                               const float* __restrict__ s0v,
                                               const int* __restrict__ count,
                                               float* __restrict__ stats,
                                               float* __restrict__ pbuf) {
    __shared__ __align__(16) float  lds_gf[CK_ * 17];   // fp32 [c][k], stride 17 (TDM pad)
    __shared__ __align__(16) __bf16 ldsB[16 * LB_];     // bf16 [k][c], stride 392
    __shared__ float sred[16 * 16];
    __shared__ float gacc[G_ * 2];

    int t = threadIdx.x;
    int b = blockIdx.x / N_, n = blockIdx.x % N_;
    const float* g0 = gf + (((size_t)b * CK_) * N_ + n) * K_;

    if (t < 32) {                                 // wave 0 issues the DMA
        tdm_load_tile(g0, (unsigned)(size_t)&lds_gf[0]);
        __builtin_amdgcn_s_wait_tensorcnt(0);
    }
    if (t < G_ * 2) gacc[t] = 0.f;
    asm volatile("" ::: "memory");                // TDM wrote LDS behind compiler's back
    __syncthreads();

    {   // one-time fp32 [c][k] -> bf16 [k][c] conversion (reused by 24 o-tiles)
        int k = t & 15, cg = t >> 4;
        for (int it = 0; it < CK_ / 16; ++it) {
            int c = it * 16 + cg;
            ldsB[k * LB_ + c] = (__bf16)lds_gf[c * 17 + k];
        }
    }
    __syncthreads();

    // ---- scores partials: thread t -> (k = t&15, c-chunk = t>>4 of 24) ----
    {
        int k = t & 15, chunk = t >> 4;
        const __bf16* qrow = qkb + ((size_t)b * N_ + n) * CK_;
        float part = 0.f;
        for (int i = 0; i < CK_ / 16; ++i) {
            int c = chunk * (CK_ / 16) + i;
            part += (float)qrow[c] * (float)ldsB[k * LB_ + c];
        }
        sred[chunk * 16 + k] = part;
    }
    __syncthreads();

    if (t < 32) {                 // wave 0: finish scores + masked softmax
        int k = t;
        if (k < 16) {
            float sc = 0.f;
            for (int ch = 0; ch < 16; ++ch) sc += sred[ch * 16 + k];
            sc = (sc + s0v[(size_t)b * N_ + n]) * SCALE_;
            int cnt = count[(size_t)b * N_ + n];
            cnt = (cnt < 1) ? 1 : (cnt > K_ ? K_ : cnt);
            if (k >= cnt) sc = -1e9f;
            float mx = sc;
#pragma unroll
            for (int m = 8; m >= 1; m >>= 1) mx = fmaxf(mx, __shfl_xor(mx, m, 16));
            float e = __expf(sc - mx);
            float ss = red16(e);
            pbuf[((size_t)b * N_ + n) * K_ + k] = e / ss;
        }
    }

    // ---- v projection (WMMA) + GN statistics ----
    int lane = t & 31, wave = t >> 5;
    int half = lane >> 4, lrow = lane & 15;
    const __bf16* bbase = &ldsB[lrow * LB_ + half * 8];
    for (int tj = 0; tj < 3; ++tj) {
        int o0 = (wave * 3 + tj) * 16;            // 8 waves x 3 tiles = 24 o-tiles
        const __bf16* abase = Wvbf + (size_t)(o0 + lrow) * CK_ + half * 8;
        v8f acc = {};
#pragma unroll 2
        for (int cs = 0; cs < CK_ / 32; ++cs) {
            v16bf a  = pack16(abase + cs * 32);
            v16bf bb = pack16(bbase + cs * 32);
            acc = __builtin_amdgcn_wmma_f32_16x16x32_bf16(false, a, false, bb,
                                                          (short)0, acc, false, false);
        }
#pragma unroll
        for (int r = 0; r < 8; ++r) {
            int o   = o0 + r + half * 8;
            float v = acc[r] + bv[o];
            float s  = red16(v);
            float q2 = red16(v * v);
            if (lrow == 0) {
                int g = o / CPG_;
                atomicAdd(&gacc[g * 2 + 0], s);
                atomicAdd(&gacc[g * 2 + 1], q2);
            }
        }
    }
    __syncthreads();
    if (t < G_ * 2) atomicAdd(&stats[(size_t)b * G_ * 2 + t], gacc[t]);
}

// ---------------------------------------------------------------------------
// Kernel 5: finalize GN stats -> mu, rsqrt(var+eps)
// ---------------------------------------------------------------------------
__global__ void k_fin(const float* __restrict__ stats, float* __restrict__ murs) {
    int i = blockIdx.x * blockDim.x + threadIdx.x;
    if (i >= B_ * G_) return;
    const float cnt = (float)(CPG_ * N_ * K_);
    float s   = stats[i * 2 + 0];
    float q   = stats[i * 2 + 1];
    float mu  = s / cnt;
    float var = q / cnt - mu * mu;
    murs[i * 2 + 0] = mu;
    murs[i * 2 + 1] = rsqrtf(var + EPS_);
}

// ---------------------------------------------------------------------------
// Kernel 6: per (b,n): TDM-stage gf tile, recompute v via WMMA, GN+ReLU,
// out[b,o,n] = K * sum_k p_k * vn[o,k]
// ---------------------------------------------------------------------------
__global__ __launch_bounds__(256) void k_out(const float* __restrict__ gf,
                                             const __bf16* __restrict__ Wvbf,
                                             const float* __restrict__ bv,
                                             const float* __restrict__ gn_w,
                                             const float* __restrict__ gn_b,
                                             const float* __restrict__ murs,
                                             const float* __restrict__ pbuf,
                                             float* __restrict__ out) {
    __shared__ __align__(16) float  lds_gf[CK_ * 17];
    __shared__ __align__(16) __bf16 ldsB[16 * LB_];
    __shared__ float psh[K_];
    int t = threadIdx.x;
    int b = blockIdx.x / N_, n = blockIdx.x % N_;
    const float* g0 = gf + (((size_t)b * CK_) * N_ + n) * K_;

    if (t < 32) {
        tdm_load_tile(g0, (unsigned)(size_t)&lds_gf[0]);
        __builtin_amdgcn_s_wait_tensorcnt(0);
    }
    if (t < K_) psh[t] = pbuf[((size_t)b * N_ + n) * K_ + t];
    asm volatile("" ::: "memory");
    __syncthreads();

    {
        int k = t & 15, cg = t >> 4;
        for (int it = 0; it < CK_ / 16; ++it) {
            int c = it * 16 + cg;
            ldsB[k * LB_ + c] = (__bf16)lds_gf[c * 17 + k];
        }
    }
    __syncthreads();

    int lane = t & 31, wave = t >> 5;
    int half = lane >> 4, lrow = lane & 15;
    const __bf16* bbase = &ldsB[lrow * LB_ + half * 8];
    for (int tj = 0; tj < 3; ++tj) {
        int o0 = (wave * 3 + tj) * 16;
        const __bf16* abase = Wvbf + (size_t)(o0 + lrow) * CK_ + half * 8;
        v8f acc = {};
#pragma unroll 2
        for (int cs = 0; cs < CK_ / 32; ++cs) {
            v16bf a  = pack16(abase + cs * 32);
            v16bf bb = pack16(bbase + cs * 32);
            acc = __builtin_amdgcn_wmma_f32_16x16x32_bf16(false, a, false, bb,
                                                          (short)0, acc, false, false);
        }
#pragma unroll
        for (int r = 0; r < 8; ++r) {
            int o = o0 + r + half * 8;
            int g = o / CPG_;
            float mu = murs[((size_t)b * G_ + g) * 2 + 0];
            float rs = murs[((size_t)b * G_ + g) * 2 + 1];
            float v  = acc[r] + bv[o];
            float vn = (v - mu) * rs * gn_w[o] + gn_b[o];
            vn = fmaxf(vn, 0.f);
            float c = red16(vn * psh[lrow]);
            if (lrow == 0)
                out[((size_t)b * COUT_ + o) * N_ + n] = (float)K_ * c;
        }
    }
}

// ---------------------------------------------------------------------------
// Host launcher
// ---------------------------------------------------------------------------
extern "C" void kernel_launch(void* const* d_in, const int* in_sizes, int n_in,
                              void* d_out, int out_size, void* d_ws, size_t ws_size,
                              hipStream_t stream) {
    (void)in_sizes; (void)n_in; (void)out_size; (void)ws_size;
    const float* feat = (const float*)d_in[0];
    const float* gf   = (const float*)d_in[1];
    const int*   cnt  = (const int*)  d_in[2];
    const float* Wq   = (const float*)d_in[3];
    const float* bq   = (const float*)d_in[4];
    const float* Wk   = (const float*)d_in[5];
    const float* bk   = (const float*)d_in[6];
    const float* Wv   = (const float*)d_in[7];
    const float* bv   = (const float*)d_in[8];
    const float* gnw  = (const float*)d_in[9];
    const float* gnb  = (const float*)d_in[10];
    float* out = (float*)d_out;

    char* ws = (char*)d_ws;
    __bf16* Mt    = (__bf16*)(ws + 0);          // 294912 B
    __bf16* Wvbf  = (__bf16*)(ws + 294912);     // 294912 B
    float*  bqk   = (float*)(ws + 589824);      // 1536 B
    float*  wqb   = (float*)(ws + 591360);      // 1536 B
    float*  cbb   = (float*)(ws + 592896);      // 4 B (padded to 256)
    float*  stats = (float*)(ws + 593152);      // 1024 B
    float*  murs  = (float*)(ws + 594176);      // 1024 B
    float*  s0    = (float*)(ws + 595200);      // 32768 B
    float*  pbuf  = (float*)(ws + 627968);      // 524288 B
    __bf16* qkb   = (__bf16*)(ws + 1152256);    // 6291456 B
    __bf16* featT = (__bf16*)(ws + 7443712);    // 6291456 B (total ~13.1 MB)

    k_prep_M   <<<(CQ_*CK_ + 255) / 256, 256, 0, stream>>>(Wq, Wk, Mt);
    k_prep_misc<<<(COUT_*CK_ + 255) / 256, 256, 0, stream>>>(Wq, Wk, Wv, bq, bk,
                                                             Wvbf, bqk, wqb, cbb, stats);
    k_prep_featT<<<(B_*N_*CQ_ + 255) / 256, 256, 0, stream>>>(feat, featT);
    {   // B * (N/16) * (CK/16) waves, 8 waves per block
        int waves = B_ * (N_ / 16) * (CK_ / 16);
        k_qk<<<waves / 8, 256, 0, stream>>>(featT, Mt, bqk, qkb);
    }
    k_s0<<<(B_*N_ + 255) / 256, 256, 0, stream>>>(feat, wqb, cbb, s0);
    k_stats<<<B_ * N_, 256, 0, stream>>>(gf, Wvbf, bv, qkb, s0, cnt, stats, pbuf);
    k_fin<<<(B_*G_ + 127) / 128, 128, 0, stream>>>(stats, murs);
    k_out<<<B_ * N_, 256, 0, stream>>>(gf, Wvbf, bv, gnw, gnb, murs, pbuf, out);
}
